// Qwen3Attention_59287728554222
// MI455X (gfx1250) — matrix operand order repaired and checked by reading, not compiled, not run on hardware
//
#include <hip/hip_runtime.h>
#include <cstdint>
#include <cstddef>

// ---------------------------------------------------------------------------
// Qwen3 attention (B=2, T=2048, H=2048, nh=16, nkv=8, d=128) for gfx1250.
// bf16 WMMA (v_wmma_f32_16x16x32_bf16) everywhere; LDS tiles double-buffered
// and staged with global_load_async_to_lds_b128 (ASYNCcnt) so DMA overlaps
// the matrix math; fragments loaded as ds_load_b128 pairs per the wave32
// WMMA VGPR layouts.
// ---------------------------------------------------------------------------

typedef __attribute__((ext_vector_type(8)))  float  v8f;
typedef __attribute__((ext_vector_type(16))) __bf16 v16bf;
typedef __attribute__((ext_vector_type(8)))  __bf16 bf16x8;
typedef __attribute__((ext_vector_type(4)))  __bf16 bf16x4;
typedef __attribute__((ext_vector_type(4)))  int    v4i;

#define B_    2
#define T_    2048
#define H_    2048
#define NH_   16
#define NKV_  8
#define D_    128
#define M_    (B_ * T_)             // 4096
#define SCALE_ 0.08838834764831845f // 1/sqrt(128)

// ---- async global->LDS (CDNA5) --------------------------------------------
#if defined(__has_builtin)
#if __has_builtin(__builtin_amdgcn_global_load_async_to_lds_b128)
#define HAVE_ASYNC_LDS 1
#endif
#endif
#ifndef HAVE_ASYNC_LDS
#define HAVE_ASYNC_LDS 0
#endif

#if HAVE_ASYNC_LDS
typedef __attribute__((address_space(1))) v4i glb_v4i;
typedef __attribute__((address_space(3))) v4i lds_v4i;
#endif

__device__ inline void async_copy16(const __bf16* g, __bf16* l) {
#if HAVE_ASYNC_LDS
  // (global src AS1, lds dst AS3, imm offset, imm cpol); tracked by ASYNCcnt
  __builtin_amdgcn_global_load_async_to_lds_b128(
      (glb_v4i*)(uintptr_t)g,
      (lds_v4i*)(uint32_t)(uintptr_t)l, 0, 0);
#else
  *(bf16x8*)l = *(const bf16x8*)g;
#endif
}

__device__ inline void async_wait0() {
#if HAVE_ASYNC_LDS
#if __has_builtin(__builtin_amdgcn_s_wait_asynccnt)
  __builtin_amdgcn_s_wait_asynccnt(0);
#else
  asm volatile("s_wait_asynccnt 0x0" ::: "memory");
#endif
#endif
}

// ---------------------------------------------------------------------------
__device__ inline v8f zero8() {
  v8f z;
#pragma unroll
  for (int i = 0; i < 8; ++i) z[i] = 0.0f;
  return z;
}

__device__ inline v8f wmma_bf16(v16bf a, v16bf b, v8f c) {
  return __builtin_amdgcn_wmma_f32_16x16x32_bf16(false, a, false, b,
                                                 (short)0, c, false, false);
}

__device__ inline v16bf concat8(bf16x8 lo, bf16x8 hi) {
  return __builtin_shufflevector(lo, hi, 0, 1, 2, 3, 4, 5, 6, 7,
                                 8, 9, 10, 11, 12, 13, 14, 15);
}

// A fragment (16x32 bf16): lane<16 -> row=lane, K {0..7}U{16..23};
// lane>=16 -> row=lane-16, K {8..15}U{24..31}. Two contiguous b128 loads.
__device__ inline v16bf load_a_frag(const __bf16* base, int row0, int k0,
                                    int ld, int lane) {
  const int row = row0 + (lane & 15);
  const int kb  = k0 + ((lane >> 4) << 3);
  const __bf16* p = base + (size_t)row * ld + kb;
  return concat8(*(const bf16x8*)p, *(const bf16x8*)(p + 16));
}

// B fragment (32x16) from matrix stored [n][k]: lane<16 -> col=lane, K {0..7}U{16..23};
// lane>=16 -> col=lane-16, K {8..15}U{24..31}. Two contiguous b128 loads.
__device__ inline v16bf load_b_frag_rowk(const __bf16* base, int col0, int k0,
                                         int ld, int lane) {
  const int col = col0 + (lane & 15);
  const int kb  = k0 + ((lane >> 4) << 3);
  const __bf16* p = base + (size_t)col * ld + kb;
  return concat8(*(const bf16x8*)p, *(const bf16x8*)(p + 16));
}

// ---------------------------------------------------------------------------
__global__ __launch_bounds__(256, 1)
void cvt_f32_bf16_kernel(const float* __restrict__ in, __bf16* __restrict__ out,
                         int n4) {
  int i = blockIdx.x * blockDim.x + threadIdx.x;
  if (i < n4) {
    const float4 v = ((const float4*)in)[i];
    bf16x4 o;
    o[0] = (__bf16)v.x; o[1] = (__bf16)v.y; o[2] = (__bf16)v.z; o[3] = (__bf16)v.w;
    ((bf16x4*)out)[i] = o;
  }
}

// ---------------------------------------------------------------------------
// Tiled GEMM: out[m][n] = sum_k A[m][k] * W[n][k]  (A @ W^T)
// 128x128 tile / block, 8 waves, K staged 32-wide, double-buffered async LDS.
// LDS layout: [sA0 | sB0 | sA1 | sB1 | sO]; buf b -> base + b*8192 elems.
// mode 0: f32 store [M x Ntot]; mode 1: bf16 repack [B,nHeads,T,D];
// mode 2: RMSNorm + RoPE -> bf16 [B,nHeads,T,D].
__global__ __launch_bounds__(256, 1)
void gemm_qkv_wmma(const __bf16* __restrict__ A, const __bf16* __restrict__ W,
                   int K, int Ntot,
                   float* __restrict__ outF, __bf16* __restrict__ outB,
                   int mode, int nHeads,
                   const float* __restrict__ normW,
                   const float* __restrict__ cosT,
                   const float* __restrict__ sinT,
                   const int* __restrict__ posIds) {
  extern __shared__ char smem[];
  __bf16* lds = (__bf16*)smem;
  float*  sO  = (float*)(lds + 4 * 128 * 32);  // 128 x 128

  const int tid  = threadIdx.x;
  const int lane = tid & 31;
  const int wv   = tid >> 5;
  const int mBase = blockIdx.x * 128;
  const int nBase = blockIdx.y * 128;
  const int rowBase = (wv & 3) * 32;   // wave sub-tile: 32 rows x 64 cols
  const int colBase = (wv >> 2) * 64;
  const int half = lane >> 4;

  // buf -> LDS offsets (computed, never a constant pointer aggregate)
  auto bufA = [&](int buf) { return lds + buf * (2 * 128 * 32); };
  auto bufB = [&](int buf) { return lds + buf * (2 * 128 * 32) + 128 * 32; };

  auto stage = [&](int k0, int buf) {
    __bf16* dA = bufA(buf);
    __bf16* dB = bufB(buf);
    for (int c = tid; c < 512; c += 256) {
      const int r  = c >> 2;
      const int cc = (c & 3) << 3;
      async_copy16(A + (size_t)(mBase + r) * K + k0 + cc, dA + r * 32 + cc);
      async_copy16(W + (size_t)(nBase + r) * K + k0 + cc, dB + r * 32 + cc);
    }
  };

  v8f acc[2][4];
#pragma unroll
  for (int s = 0; s < 2; ++s)
#pragma unroll
    for (int c = 0; c < 4; ++c) acc[s][c] = zero8();

  stage(0, 0);  // prologue: tile 0 in flight

  for (int k0 = 0; k0 < K; k0 += 32) {
    const int cur = (k0 >> 5) & 1;
    async_wait0();      // current tile's DMA done (this wave)
    __syncthreads();    // all waves see it; alternate buffer is free
    if (k0 + 32 < K) stage(k0 + 32, cur ^ 1);  // next tile's DMA overlaps WMMA
    if (k0 + 64 < K) {  // warm L2 for the tile after next
      if (tid < 128)
        __builtin_prefetch(A + (size_t)(mBase + tid) * K + k0 + 64, 0, 3);
      else
        __builtin_prefetch(W + (size_t)(nBase + tid - 128) * K + k0 + 64, 0, 3);
    }

    const __bf16* cA = bufA(cur);
    const __bf16* cB = bufB(cur);
    v16bf af[2], bf[4];
#pragma unroll
    for (int s = 0; s < 2; ++s) af[s] = load_a_frag(cA, rowBase + s * 16, 0, 32, lane);
#pragma unroll
    for (int c = 0; c < 4; ++c) bf[c] = load_b_frag_rowk(cB, colBase + c * 16, 0, 32, lane);
#pragma unroll
    for (int s = 0; s < 2; ++s)
#pragma unroll
      for (int c = 0; c < 4; ++c) acc[s][c] = wmma_bf16(af[s], bf[c], acc[s][c]);
  }

  // spill accumulators (C layout: vgpr i -> row i + 8*half, col = lane&15)
  __syncthreads();
#pragma unroll
  for (int s = 0; s < 2; ++s)
#pragma unroll
    for (int c = 0; c < 4; ++c)
#pragma unroll
      for (int i = 0; i < 8; ++i) {
        const int r  = rowBase + s * 16 + i + (half << 3);
        const int cc = colBase + c * 16 + (lane & 15);
        sO[r * 128 + cc] = acc[s][c][i];
      }
  __syncthreads();

  if (mode == 0) {
    for (int c = tid; c < 4096; c += 256) {  // float4 chunks
      const int r = c >> 5, cc = (c & 31) << 2;
      *(float4*)(outF + (size_t)(mBase + r) * Ntot + nBase + cc) =
          *(const float4*)(sO + r * 128 + cc);
    }
  } else if (mode == 1) {
    const int head = nBase >> 7;
    for (int c = tid; c < 4096; c += 256) {
      const int r = c >> 5, cc = (c & 31) << 2;
      const int m = mBase + r;
      const int bb = m >> 11, t = m & (T_ - 1);
      const float4 v = *(const float4*)(sO + r * 128 + cc);
      bf16x4 o4;
      o4[0] = (__bf16)v.x; o4[1] = (__bf16)v.y; o4[2] = (__bf16)v.z; o4[3] = (__bf16)v.w;
      *(bf16x4*)(outB + (((size_t)bb * nHeads + head) * T_ + t) * D_ + cc) = o4;
    }
  } else {
    // one row == one head's 128-dim vector: RMSNorm then RoPE
    if (tid < 128) {
      const int r = tid;
      const int m = mBase + r;
      const int bb = m >> 11, t = m & (T_ - 1);
      const int head = nBase >> 7;
      const int pos = posIds[bb * T_ + t];
      const float* rowp = sO + r * 128;
      float ss = 0.0f;
      for (int j = 0; j < 128; ++j) { const float v = rowp[j]; ss += v * v; }
      const float inv = rsqrtf(ss * (1.0f / 128.0f) + 1e-6f);
      const float* cp = cosT + (size_t)pos * 128;
      const float* sp = sinT + (size_t)pos * 128;
      __bf16* op = outB + (((size_t)bb * nHeads + head) * T_ + t) * D_;
      for (int j = 0; j < 128; ++j) {
        const float nv = rowp[j] * inv * normW[j];
        const float rot = (j < 64) ? -(rowp[j + 64] * inv * normW[j + 64])
                                   :  (rowp[j - 64] * inv * normW[j - 64]);
        op[j] = (__bf16)(nv * cp[j] + rot * sp[j]);
      }
    }
  }
}

// ---------------------------------------------------------------------------
// Flash attention: one block per (qTile, head, batch). 8 waves; each wave owns
// a 16-row query stripe and all 128 keys of the current K tile. K/V tiles are
// double-buffered so the next tile's DMA overlaps WMMA + softmax.
// LDS layout (16384-elem tiles): [sQ | sK0 | sK1 | sVt0 | sVt1 | sP]
__global__ __launch_bounds__(256, 1)
void flash_attn_wmma(const __bf16* __restrict__ Q,   // [B, NH, T, D]
                     const __bf16* __restrict__ Kk,  // [B, NKV, T, D]
                     const __bf16* __restrict__ V,   // [B, NKV, T, D]
                     __bf16* __restrict__ attnOut) { // [B, T, NH*D]
  extern __shared__ char smem[];
  __bf16* lds = (__bf16*)smem;
  __bf16* sQ  = lds;                    // [row][d]
  __bf16* sP  = lds + 5 * 128 * 128;    // [row][key]

  const int tid  = threadIdx.x;
  const int lane = tid & 31;
  const int wv   = tid >> 5;
  const int half = lane >> 4;
  const int qt = blockIdx.x;
  const int h  = blockIdx.y;
  const int b  = blockIdx.z;
  const int qBase = qt * 128;
  const int hkv = h / (NH_ / NKV_);

  const __bf16* gQ = Q  + ((size_t)(b * NH_  + h  ) * T_ + qBase) * D_;
  const __bf16* gK = Kk + ((size_t)(b * NKV_ + hkv) * T_) * D_;
  const __bf16* gV = V  + ((size_t)(b * NKV_ + hkv) * T_) * D_;

  auto bufK = [&](int buf) { return lds + (1 + buf) * (128 * 128); };  // [key][d]
  auto bufV = [&](int buf) { return lds + (3 + buf) * (128 * 128); };  // [d][key]

  auto stageKV = [&](int kt, int buf) {
    __bf16* dK  = bufK(buf);
    __bf16* dVt = bufV(buf);
    const __bf16* gKt = gK + (size_t)kt * 128 * D_;
    const __bf16* gVt = gV + (size_t)kt * 128 * D_;
    for (int c = tid; c < 2048; c += 256) {  // K: async straight copy
      const int r = c >> 4, cc = (c & 15) << 3;
      async_copy16(gKt + (size_t)r * D_ + cc, dK + r * 128 + cc);
    }
    for (int c = tid; c < 2048; c += 256) {  // V: load rows, store transposed
      const int key = c >> 4, d0 = (c & 15) << 3;
      const bf16x8 v = *(const bf16x8*)(gVt + (size_t)key * D_ + d0);
#pragma unroll
      for (int j = 0; j < 8; ++j) dVt[(d0 + j) * 128 + key] = v[j];
    }
  };

  // prologue: Q tile + first K/V tile in flight
  for (int c = tid; c < 2048; c += 256) {
    const int r = c >> 4, cc = (c & 15) << 3;
    async_copy16(gQ + (size_t)r * D_ + cc, sQ + r * 128 + cc);
  }
  stageKV(0, 0);

  float m_i[8], l_i[8];
  v8f o[8];
#pragma unroll
  for (int i = 0; i < 8; ++i) { m_i[i] = -1e30f; l_i[i] = 0.0f; o[i] = zero8(); }

  for (int kt = 0; kt <= qt; ++kt) {
    const int cur = kt & 1;
    async_wait0();      // this tile's async DMA done (per-wave)
    __syncthreads();    // everyone sees it; alternate buffer is free
    if (kt + 1 <= qt) stageKV(kt + 1, cur ^ 1);  // next tile DMA overlaps math
    const __bf16* sK  = bufK(cur);
    const __bf16* sVt = bufV(cur);

    // S = Q K^T over d=128 (4 WMMA k-steps) x 8 key-column tiles
    v8f st[8];
#pragma unroll
    for (int ct = 0; ct < 8; ++ct) st[ct] = zero8();
#pragma unroll
    for (int ks = 0; ks < 4; ++ks) {
      const v16bf aq = load_a_frag(sQ, wv * 16, ks * 32, 128, lane);
#pragma unroll
      for (int ct = 0; ct < 8; ++ct) {
        const v16bf bk = load_b_frag_rowk(sK, ct * 16, ks * 32, 128, lane);
        st[ct] = wmma_bf16(aq, bk, st[ct]);
      }
    }

    // scale, causal mask (diagonal tile only), online softmax per row
#pragma unroll
    for (int i = 0; i < 8; ++i) {
      const int row = qBase + wv * 16 + i + (half << 3);
      float mx = -1e30f;
#pragma unroll
      for (int ct = 0; ct < 8; ++ct) {
        float s = st[ct][i] * SCALE_;
        if (kt == qt) {
          const int key = kt * 128 + ct * 16 + (lane & 15);
          if (key > row) s = -1e30f;
        }
        st[ct][i] = s;
        mx = fmaxf(mx, s);
      }
#pragma unroll
      for (int d = 1; d < 16; d <<= 1) mx = fmaxf(mx, __shfl_xor(mx, d, 32));
      const float mnew  = fmaxf(m_i[i], mx);
      const float alpha = __expf(m_i[i] - mnew);
      float rsum = 0.0f;
#pragma unroll
      for (int ct = 0; ct < 8; ++ct) {
        const float p = __expf(st[ct][i] - mnew);
        st[ct][i] = p;
        rsum += p;
      }
#pragma unroll
      for (int d = 1; d < 16; d <<= 1) rsum += __shfl_xor(rsum, d, 32);
      l_i[i] = l_i[i] * alpha + rsum;
      m_i[i] = mnew;
#pragma unroll
      for (int ct = 0; ct < 8; ++ct) o[ct][i] *= alpha;
    }

    // C-layout P -> LDS (A-layout source for the O-update WMMA)
#pragma unroll
    for (int ct = 0; ct < 8; ++ct)
#pragma unroll
      for (int i = 0; i < 8; ++i) {
        const int r  = wv * 16 + i + (half << 3);
        const int cc = ct * 16 + (lane & 15);
        sP[r * 128 + cc] = (__bf16)st[ct][i];
      }
    __syncthreads();

    // O += P V : k = 128 keys (4 WMMA k-steps) x 8 d-column tiles
#pragma unroll
    for (int ks = 0; ks < 4; ++ks) {
      const v16bf ap = load_a_frag(sP, wv * 16, ks * 32, 128, lane);
#pragma unroll
      for (int ct = 0; ct < 8; ++ct) {
        const v16bf bv = load_b_frag_rowk(sVt, ct * 16, ks * 32, 128, lane);
        o[ct] = wmma_bf16(ap, bv, o[ct]);
      }
    }
  }

  // normalize by l and write [B, T, NH*D] bf16
#pragma unroll
  for (int ct = 0; ct < 8; ++ct)
#pragma unroll
    for (int i = 0; i < 8; ++i) {
      const int row = qBase + wv * 16 + i + (half << 3);
      const int col = h * D_ + ct * 16 + (lane & 15);
      attnOut[((size_t)b * T_ + row) * (NH_ * D_) + col] = (__bf16)(o[ct][i] / l_i[i]);
    }
}

// ---------------------------------------------------------------------------
extern "C" void kernel_launch(void* const* d_in, const int* in_sizes, int n_in,
                              void* d_out, int out_size, void* d_ws, size_t ws_size,
                              hipStream_t stream) {
  (void)in_sizes; (void)n_in; (void)out_size; (void)ws_size;

  const float* x     = (const float*)d_in[0];
  const float* cosT  = (const float*)d_in[1];
  const float* sinT  = (const float*)d_in[2];
  const int*   pids  = (const int*)d_in[3];
  const float* Wq    = (const float*)d_in[4];
  const float* Wk    = (const float*)d_in[5];
  const float* Wv    = (const float*)d_in[6];
  const float* Wo    = (const float*)d_in[7];
  const float* qw    = (const float*)d_in[8];
  const float* kw    = (const float*)d_in[9];
  float* out = (float*)d_out;

  char* ws = (char*)d_ws;
  __bf16* xb   = (__bf16*)ws;  ws += (size_t)M_ * H_ * 2;
  __bf16* wqb  = (__bf16*)ws;  ws += (size_t)(NH_ * D_) * H_ * 2;
  __bf16* wkb  = (__bf16*)ws;  ws += (size_t)(NKV_ * D_) * H_ * 2;
  __bf16* wvb  = (__bf16*)ws;  ws += (size_t)(NKV_ * D_) * H_ * 2;
  __bf16* wob  = (__bf16*)ws;  ws += (size_t)H_ * (NH_ * D_) * 2;
  __bf16* qbuf = (__bf16*)ws;  ws += (size_t)B_ * NH_ * T_ * D_ * 2;
  __bf16* kbuf = (__bf16*)ws;  ws += (size_t)B_ * NKV_ * T_ * D_ * 2;
  __bf16* vbuf = (__bf16*)ws;  ws += (size_t)B_ * NKV_ * T_ * D_ * 2;
  __bf16* attn = (__bf16*)ws;  ws += (size_t)M_ * (NH_ * D_) * 2;

  const size_t gemmSmem  = 4 * 128 * 32 * 2 + 128 * 128 * 4;  // 96 KB
  const size_t flashSmem = (size_t)6 * 128 * 128 * 2;         // 192 KB
  (void)hipFuncSetAttribute(reinterpret_cast<const void*>(&gemm_qkv_wmma),
                            hipFuncAttributeMaxDynamicSharedMemorySize, (int)gemmSmem);
  (void)hipFuncSetAttribute(reinterpret_cast<const void*>(&flash_attn_wmma),
                            hipFuncAttributeMaxDynamicSharedMemorySize, (int)flashSmem);

  auto cvt = [&](const float* src, __bf16* dst, int n) {
    const int n4 = n / 4;
    cvt_f32_bf16_kernel<<<(n4 + 255) / 256, 256, 0, stream>>>(src, dst, n4);
  };
  cvt(x,  xb,  M_ * H_);
  cvt(Wq, wqb, NH_ * D_ * H_);
  cvt(Wk, wkb, NKV_ * D_ * H_);
  cvt(Wv, wvb, NKV_ * D_ * H_);
  cvt(Wo, wob, H_ * NH_ * D_);

  gemm_qkv_wmma<<<dim3(M_ / 128, (NH_ * D_) / 128), 256, gemmSmem, stream>>>(
      xb, wqb, H_, 0, nullptr, qbuf, /*mode=*/2, NH_, qw, cosT, sinT, pids);
  gemm_qkv_wmma<<<dim3(M_ / 128, (NKV_ * D_) / 128), 256, gemmSmem, stream>>>(
      xb, wkb, H_, 0, nullptr, kbuf, /*mode=*/2, NKV_, kw, cosT, sinT, pids);
  gemm_qkv_wmma<<<dim3(M_ / 128, (NKV_ * D_) / 128), 256, gemmSmem, stream>>>(
      xb, wvb, H_, 0, nullptr, vbuf, /*mode=*/1, NKV_, nullptr, nullptr, nullptr, nullptr);

  flash_attn_wmma<<<dim3(T_ / 128, NH_, B_), 256, flashSmem, stream>>>(
      qbuf, kbuf, vbuf, attn);

  gemm_qkv_wmma<<<dim3(M_ / 128, H_ / 128), 256, gemmSmem, stream>>>(
      attn, wob, NH_ * D_, H_, out, nullptr, /*mode=*/0, 0,
      nullptr, nullptr, nullptr, nullptr);
}
